// ViT_56977036148780
// MI455X (gfx1250) — matrix operand order repaired and checked by reading, not compiled, not run on hardware
//
#include <hip/hip_runtime.h>
#include <hip/hip_bf16.h>
#include <type_traits>

typedef unsigned short u16;
typedef __attribute__((ext_vector_type(16))) __bf16 v16bf;
typedef __attribute__((ext_vector_type(8)))  float  v8f;
typedef __attribute__((ext_vector_type(4)))  unsigned int v4u;
typedef __attribute__((ext_vector_type(2)))  unsigned int v2u;

#define VB   32
#define VD   768
#define VNH  12
#define VDH  64
#define VL   12
#define VT   197          // tokens per image (196 patches + cls)
#define VM   (VB*VT)      // 6304 total tokens
#define VMLP 3072
#define VNP  196
#define VNPAT (VB*VNP)    // 6272
#define SLD  208          // padded leading dim for attention matrices

__device__ __forceinline__ u16 f2bf(float f) {
    unsigned u = __float_as_uint(f);
    unsigned r = u + 0x7FFFu + ((u >> 16) & 1u);   // round-to-nearest-even
    return (u16)(r >> 16);
}

union FragAB { v4u q[2]; v16bf v; };

// ---------------------------------------------------------------------------
// bf16 WMMA GEMM:  C[M,N] = epilogue( alpha * A[M,K] @ Bt[N,K]^T )
// A and Bt bf16 row-major (Bt is B stored column-major, rows have K extent).
// Interior tiles: global_load_async_to_lds_b128, double-buffered LDS,
//   early-issued next tile with s_wait_asynccnt threshold (in-order done),
//   guard-free epilogue.
// Edge tiles / K tail: clamped unconditional loads + mask (no exec branches),
//   guarded epilogue.
// Epilogue duplicated per path (early return) to avoid acc spills at joins.
// Block 256 threads (8 waves, 4x2 grid). TN = column subtiles per wave:
//   TN=2: tile 128x64, wave 32x32, 4 wmma/step
//   TN=4: tile 128x128, wave 32x64, 8 wmma/step
// ---------------------------------------------------------------------------
template <int TN>
__global__ __launch_bounds__(256) void gemm_bf16_wmma(
    const u16* __restrict__ A, int lda, long long sA,
    const u16* __restrict__ Bt, int ldb, long long sB,
    const float* __restrict__ bias,
    const float* __restrict__ resid,
    float* __restrict__ Cf, u16* __restrict__ Ch,
    int ldc, int cDiv, long long cOuter, long long cInner,
    int M, int N, int K, float alpha, int doGelu)
{
    constexpr int BN = 32 * TN;                 // 64 or 128
    __shared__ u16 As[2][128 * 40];             // [buf][m*40 + k], 80B row (skewed)
    __shared__ u16 Bs[2][BN * 40];              // [buf][n*40 + k]

    const int tid = threadIdx.x;
    const int z   = blockIdx.z;
    A  += (long long)z * sA;
    Bt += (long long)z * sB;
    const long long cOff = (long long)(z / cDiv) * cOuter + (long long)(z % cDiv) * cInner;

    const int m0 = blockIdx.y * 128;
    const int n0 = blockIdx.x * BN;

    const int wid  = tid >> 5, lane = tid & 31;
    const int wm   = wid >> 1, wn = wid & 1;
    const int mlo  = lane & 15;
    const int koffA = (lane >> 4) << 3;   // 0 or 8
    const int koffB = (lane >> 4) << 3;   // 0 or 8

    v8f acc[2][TN];
    for (int s = 0; s < 2; ++s)
        for (int t = 0; t < TN; ++t)
            for (int i = 0; i < 8; ++i) acc[s][t][i] = 0.f;

    // thread -> 16B chunk coords (async path)
    const int rowC = tid >> 2;                 // 0..63
    const int kC   = (tid & 3) << 3;           // 0,8,16,24

    auto issueAsync = [&](int buf, int k0) {
        #pragma unroll
        for (int cb = 0; cb < 2; ++cb) {       // A: 128 rows
            unsigned lds = (unsigned)(size_t)&As[buf][(rowC + cb * 64) * 40 + kC];
            const u16* g = A + (long long)(m0 + rowC + cb * 64) * lda + (k0 + kC);
            asm volatile("global_load_async_to_lds_b128 %0, %1, off"
                         :: "v"(lds), "v"((unsigned long long)(size_t)g) : "memory");
        }
        #pragma unroll
        for (int cb = 0; cb < TN / 2; ++cb) {  // B: BN rows
            unsigned lds = (unsigned)(size_t)&Bs[buf][(rowC + cb * 64) * 40 + kC];
            const u16* g = Bt + (long long)(n0 + rowC + cb * 64) * ldb + (k0 + kC);
            asm volatile("global_load_async_to_lds_b128 %0, %1, off"
                         :: "v"(lds), "v"((unsigned long long)(size_t)g) : "memory");
        }
    };

    // Edge staging: clamped UNCONDITIONAL loads + arithmetic masking.
    auto stageGuard = [&](int buf, int k0) {
        #pragma unroll
        for (int it = 0; it < 4; ++it) {              // A: 128x32, 4 bf16/thread/iter
            int c = tid + it * 256;
            int m = c >> 3;
            int k = (c & 7) << 2;
            int gm = m0 + m, gk = k0 + k;
            int cm = (gm < M) ? gm : (M - 1);
            const u16* rowp = A + (long long)cm * lda;
            unsigned rowMask = (gm < M) ? 0xFFFFFFFFu : 0u;
            unsigned w[2];
            #pragma unroll
            for (int h = 0; h < 2; ++h) {
                int gk0 = gk + 2 * h, gk1 = gk0 + 1;
                int ck0 = (gk0 < K) ? gk0 : (K - 1);
                int ck1 = (gk1 < K) ? gk1 : (K - 1);
                unsigned mk0 = (gk0 < K) ? 0xFFFFu : 0u;
                unsigned mk1 = (gk1 < K) ? 0xFFFFu : 0u;
                unsigned lo = (unsigned)rowp[ck0] & mk0;
                unsigned hi = (unsigned)rowp[ck1] & mk1;
                w[h] = (lo | (hi << 16)) & rowMask;
            }
            v2u val; val.x = w[0]; val.y = w[1];
            *(v2u*)(&As[buf][m * 40 + k]) = val;
        }
        #pragma unroll
        for (int it = 0; it < TN; ++it) {             // B: BNx32, 4 bf16/thread/iter
            int c = tid + it * 256;
            int n = c >> 3;
            int k = (c & 7) << 2;
            int gn = n0 + n, gk = k0 + k;
            int cn = (gn < N) ? gn : (N - 1);
            const u16* rowp = Bt + (long long)cn * ldb;
            unsigned rowMask = (gn < N) ? 0xFFFFFFFFu : 0u;
            unsigned w[2];
            #pragma unroll
            for (int h = 0; h < 2; ++h) {
                int gk0 = gk + 2 * h, gk1 = gk0 + 1;
                int ck0 = (gk0 < K) ? gk0 : (K - 1);
                int ck1 = (gk1 < K) ? gk1 : (K - 1);
                unsigned mk0 = (gk0 < K) ? 0xFFFFu : 0u;
                unsigned mk1 = (gk1 < K) ? 0xFFFFu : 0u;
                unsigned lo = (unsigned)rowp[ck0] & mk0;
                unsigned hi = (unsigned)rowp[ck1] & mk1;
                w[h] = (lo | (hi << 16)) & rowMask;
            }
            v2u val; val.x = w[0]; val.y = w[1];
            *(v2u*)(&Bs[buf][n * 40 + k]) = val;
        }
    };

    auto compute = [&](int buf) {
        FragAB af[2], bfr[TN];
        #pragma unroll
        for (int s = 0; s < 2; ++s) {
            int rb = wm * 32 + s * 16 + mlo;
            af[s].q[0] = *(const v4u*)(&As[buf][rb * 40 + koffA]);
            af[s].q[1] = *(const v4u*)(&As[buf][rb * 40 + 16 + koffA]);
        }
        #pragma unroll
        for (int t = 0; t < TN; ++t) {
            int nb = wn * (16 * TN) + t * 16 + mlo;
            bfr[t].q[0] = *(const v4u*)(&Bs[buf][nb * 40 + koffB]);
            bfr[t].q[1] = *(const v4u*)(&Bs[buf][nb * 40 + koffB + 16]);
        }
        #pragma unroll
        for (int s = 0; s < 2; ++s)
            #pragma unroll
            for (int t = 0; t < TN; ++t)
                acc[s][t] = __builtin_amdgcn_wmma_f32_16x16x32_bf16(
                    false, af[s].v, false, bfr[t].v, (short)0, acc[s][t], false, false);
    };

    // GUARDED=false: interior blocks, all M/N guards statically removed.
    auto epilogue = [&](auto guardedTag) {
        constexpr bool GUARDED = decltype(guardedTag)::value;
        float* CfB = Cf ? Cf + cOff : nullptr;
        u16*   ChB = Ch ? Ch + cOff : nullptr;
        const float* rB = resid ? resid + cOff : nullptr;
        const int rlo = (lane >> 4) << 3;
        #pragma unroll
        for (int s = 0; s < 2; ++s)
            #pragma unroll
            for (int t = 0; t < TN; ++t) {
                int mb = m0 + wm * 32 + s * 16 + rlo;
                int nb = n0 + wn * (16 * TN) + t * 16 + mlo;
                if (GUARDED && nb >= N) continue;
                float bv = bias ? bias[nb] : 0.f;
                #pragma unroll
                for (int r = 0; r < 8; ++r) {
                    int m = mb + r;
                    if (GUARDED && m >= M) break;
                    float f = acc[s][t][r] * alpha + bv;
                    if (doGelu) f = 0.5f * f * (1.f + erff(f * 0.70710678118654752f));
                    if (rB) f += rB[(long long)m * ldc + nb];
                    if (CfB) CfB[(long long)m * ldc + nb] = f;
                    if (ChB) ChB[(long long)m * ldc + nb] = f2bf(f);
                }
            }
    };

    const bool interior = (m0 + 128 <= M) && (n0 + BN <= N);
    const int nFull = K >> 5;
    const int tail  = K & 31;

    if (interior) {
        if (nFull > 0) {
            issueAsync(0, 0);
            for (int i = 0; i < nFull; ++i) {
                if (i + 1 < nFull) {
                    issueAsync((i + 1) & 1, (i + 1) << 5);   // buf WAR covered by prev trailing barrier
                    if constexpr (TN == 4) asm volatile("s_wait_asynccnt 0x4" ::: "memory");
                    else                   asm volatile("s_wait_asynccnt 0x3" ::: "memory");
                } else {
                    asm volatile("s_wait_asynccnt 0x0" ::: "memory");
                }
                __syncthreads();                          // buf i&1 staged for all waves
                compute(i & 1);
                __syncthreads();                          // all done reading buf i&1
            }
        }
        if (tail) {
            stageGuard(0, nFull << 5);
            __syncthreads();
            compute(0);
        }
        epilogue(std::false_type{});                      // guard-free, no join spill
        return;
    }

    for (int k0 = 0; k0 < K; k0 += 32) {
        __syncthreads();
        stageGuard(0, k0);
        __syncthreads();
        compute(0);
    }
    epilogue(std::true_type{});
}

// ---------------------------------------------------------------------------
// Elementwise / support kernels
// ---------------------------------------------------------------------------

// f32 [K,N] -> bf16 transposed [N,K]
__global__ void k_cvt_t(const float* __restrict__ in, u16* __restrict__ out,
                        int Kd, int Nd) {
    long long i = (long long)blockIdx.x * 256 + threadIdx.x;
    long long n = (long long)Kd * Nd;
    if (i >= n) return;
    int r = (int)(i / Kd);        // output row = original column
    int k = (int)(i % Kd);
    out[i] = f2bf(in[(long long)k * Nd + r]);
}

// qkv_w[l] [NH, D, 3DH] -> bf16 [NH*3DH, D] (transposed, head-major columns)
__global__ void k_cvt_qkv_t(const float* __restrict__ in, u16* __restrict__ out) {
    long long i = (long long)blockIdx.x * 256 + threadIdx.x;
    const long long n = (long long)(VNH * 3 * VDH) * VD;
    if (i >= n) return;
    int r = (int)(i / VD);        // h*192+f
    int d = (int)(i % VD);
    int h = r / (3 * VDH), f = r % (3 * VDH);
    out[i] = f2bf(in[((long long)h * VD + d) * (3 * VDH) + f]);
}

// x[B,3,224,224] -> patches bf16 [6272, 768], feature order (C,P,P)
__global__ void k_im2col(const float* __restrict__ x, u16* __restrict__ out) {
    long long i = (long long)blockIdx.x * 256 + threadIdx.x;
    const long long n = (long long)VNPAT * VD;
    if (i >= n) return;
    int r = (int)(i / VD), c = (int)(i % VD);
    int b = r / VNP, p = r % VNP;
    int py = p / 14, px = p % 14;
    int ch = c / 256, rem = c % 256;
    int ii = rem / 16, jj = rem % 16;
    out[i] = f2bf(x[(((long long)b * 3 + ch) * 224 + (py * 16 + ii)) * 224 + (px * 16 + jj)]);
}

// z[b,0,:] = cls + pos[0];  z[b,1+p,:] = proj[b*196+p] + pos[1+p]
__global__ void k_assemble(const float* __restrict__ proj, const float* __restrict__ cls,
                           const float* __restrict__ pos, float* __restrict__ z) {
    long long i = (long long)blockIdx.x * 256 + threadIdx.x;
    const long long n = (long long)VM * VD;
    if (i >= n) return;
    int row = (int)(i / VD), c = (int)(i % VD);
    int b = row / VT, t = row % VT;
    float v = (t == 0) ? cls[c] : proj[((long long)b * VNP + (t - 1)) * VD + c];
    z[i] = v + pos[(long long)t * VD + c];
}

// qkv bf16 [6304,2304] -> q [384,197,64], k [384,197,64], vT [384,64,208]
__global__ void k_split(const u16* __restrict__ qkv, u16* __restrict__ q,
                        u16* __restrict__ k, u16* __restrict__ vT) {
    long long i = (long long)blockIdx.x * 256 + threadIdx.x;
    const long long n = (long long)VM * (VNH * 3 * VDH);
    if (i >= n) return;
    int row = (int)(i / (VNH * 3 * VDH));
    int col = (int)(i % (VNH * 3 * VDH));
    int b = row / VT, t = row % VT;
    int h = col / (3 * VDH), f = col % (3 * VDH);
    int bh = b * VNH + h;
    u16 u = qkv[i];
    if (f < VDH)           q[((long long)bh * VT + t) * VDH + f]            = u;
    else if (f < 2 * VDH)  k[((long long)bh * VT + t) * VDH + (f - VDH)]    = u;
    else                   vT[(long long)bh * VDH * SLD + (long long)(f - 2 * VDH) * SLD + t] = u;
}

// LayerNorm over D=768; one block per row; out bf16 compacted by blockIdx.
__global__ __launch_bounds__(256) void k_ln(const float* __restrict__ in,
                                            const float* __restrict__ g,
                                            const float* __restrict__ b,
                                            u16* __restrict__ out, int rowStep) {
    int row = blockIdx.x * rowStep;
    const float* x = in + (long long)row * VD;
    u16* o = out + (long long)blockIdx.x * VD;
    int t = threadIdx.x;
    float v0 = x[t], v1 = x[t + 256], v2 = x[t + 512];
    __shared__ float r1[256], r2[256];
    r1[t] = v0 + v1 + v2;
    r2[t] = v0 * v0 + v1 * v1 + v2 * v2;
    __syncthreads();
    for (int off = 128; off > 0; off >>= 1) {
        if (t < off) { r1[t] += r1[t + off]; r2[t] += r2[t + off]; }
        __syncthreads();
    }
    float mean = r1[0] * (1.f / VD);
    float var  = r2[0] * (1.f / VD) - mean * mean;
    float inv  = rsqrtf(var + 1e-5f);
    o[t]       = f2bf((v0 - mean) * inv * g[t]       + b[t]);
    o[t + 256] = f2bf((v1 - mean) * inv * g[t + 256] + b[t + 256]);
    o[t + 512] = f2bf((v2 - mean) * inv * g[t + 512] + b[t + 512]);
}

// softmax over 197 cols of S (f32, ld 208) -> probs bf16 (ld 208)
__global__ __launch_bounds__(256) void k_softmax(const float* __restrict__ S,
                                                 u16* __restrict__ P) {
    long long r = blockIdx.x;
    long long zz = r / VT;
    int nrow = (int)(r % VT);
    const float* srow = S + zz * (long long)(SLD * SLD) + (long long)nrow * SLD;
    u16* prow = P + zz * (long long)(SLD * SLD) + (long long)nrow * SLD;
    int t = threadIdx.x;
    float v = (t < VT) ? srow[t] : -3.0e38f;
    __shared__ float red[256];
    red[t] = v; __syncthreads();
    for (int off = 128; off > 0; off >>= 1) {
        if (t < off) red[t] = fmaxf(red[t], red[t + off]);
        __syncthreads();
    }
    float mx = red[0]; __syncthreads();
    float e = (t < VT) ? __expf(v - mx) : 0.f;
    red[t] = e; __syncthreads();
    for (int off = 128; off > 0; off >>= 1) {
        if (t < off) red[t] += red[t + off];
        __syncthreads();
    }
    float inv = 1.f / red[0];
    if (t < VT) prow[t] = f2bf(e * inv);
}

// ---------------------------------------------------------------------------
extern "C" void kernel_launch(void* const* d_in, const int* in_sizes, int n_in,
                              void* d_out, int out_size, void* d_ws, size_t ws_size,
                              hipStream_t stream) {
    (void)in_sizes; (void)n_in; (void)out_size; (void)ws_size;
    const float* x      = (const float*)d_in[0];
    const float* proj_w = (const float*)d_in[1];
    const float* proj_b = (const float*)d_in[2];
    const float* cls    = (const float*)d_in[3];
    const float* pos    = (const float*)d_in[4];
    const float* ln1_g  = (const float*)d_in[5];
    const float* ln1_b  = (const float*)d_in[6];
    const float* qkv_w  = (const float*)d_in[7];
    const float* qkv_b  = (const float*)d_in[8];
    const float* msa_w  = (const float*)d_in[9];
    const float* msa_b  = (const float*)d_in[10];
    const float* ln2_g  = (const float*)d_in[11];
    const float* ln2_b  = (const float*)d_in[12];
    const float* w1     = (const float*)d_in[13];
    const float* b1     = (const float*)d_in[14];
    const float* w2     = (const float*)d_in[15];
    const float* b2     = (const float*)d_in[16];
    const float* lnf_g  = (const float*)d_in[17];
    const float* lnf_b  = (const float*)d_in[18];
    const float* head_w = (const float*)d_in[19];
    const float* head_b = (const float*)d_in[20];

    // ---- workspace carve ----
    char* p = (char*)d_ws;
    auto alloc = [&](long long bytes) -> char* {
        char* r = p; p += (bytes + 255) & ~255LL; return r;
    };
    const long long BH = (long long)VB * VNH;                 // 384
    float* z    = (float*)alloc((long long)VM * VD * 4);
    float* z2   = (float*)alloc((long long)VM * VD * 4);
    u16*   z1h  = (u16*)  alloc((long long)VM * VD * 2);
    u16*   qkvh = (u16*)  alloc((long long)VM * VNH * 3 * VDH * 2);
    u16*   qh   = (u16*)  alloc(BH * VT * VDH * 2);
    u16*   kh   = (u16*)  alloc(BH * VT * VDH * 2);
    u16*   vTh  = (u16*)  alloc(BH * VDH * SLD * 2);
    u16*   sah  = (u16*)  alloc((long long)VM * VD * 2);
    u16*   wh   = (u16*)  alloc((long long)VD * VMLP * 2);
    u16*   clsh = (u16*)  alloc((long long)VB * VD * 2);
    char*  big  = alloc(BH * SLD * SLD * 4 + BH * SLD * SLD * 2);
    float* S    = (float*)big;                                 // attention phase
    u16*   Ph   = (u16*)(big + BH * SLD * SLD * 4);
    float* tmpf = (float*)big;                                 // pre-loop alias
    u16*   pth  = (u16*)(big + (long long)VNPAT * VD * 4);
    u16*   h1h  = (u16*)big;                                   // MLP phase alias

    auto cdiv = [](long long a, long long b) { return (unsigned)((a + b - 1) / b); };
    auto gemm = [&](int TN, const u16* A, int lda, long long sA,
                    const u16* Bm, int ldb, long long sB,
                    const float* bias, const float* resid, float* Cf, u16* Ch,
                    int ldc, int cD, long long cO, long long cI,
                    int M, int N, int K, float alpha, int gelu, int batch) {
        dim3 g(cdiv(N, 32 * TN), cdiv(M, 128), batch);
        if (TN == 4)
            gemm_bf16_wmma<4><<<g, dim3(256), 0, stream>>>(A, lda, sA, Bm, ldb, sB, bias, resid,
                Cf, Ch, ldc, cD, cO, cI, M, N, K, alpha, gelu);
        else
            gemm_bf16_wmma<2><<<g, dim3(256), 0, stream>>>(A, lda, sA, Bm, ldb, sB, bias, resid,
                Cf, Ch, ldc, cD, cO, cI, M, N, K, alpha, gelu);
    };

    // ---- patch embedding ----
    k_im2col<<<cdiv((long long)VNPAT * VD, 256), 256, 0, stream>>>(x, pth);
    k_cvt_t<<<cdiv((long long)VD * VD, 256), 256, 0, stream>>>(proj_w, wh, VD, VD);
    gemm(4, pth, VD, 0, wh, VD, 0, proj_b, nullptr, tmpf, nullptr,
         VD, 1, 0, 0, VNPAT, VD, VD, 1.f, 0, 1);
    k_assemble<<<cdiv((long long)VM * VD, 256), 256, 0, stream>>>(tmpf, cls, pos, z);

    // ---- transformer layers ----
    const long long qkvWsz = (long long)VNH * VD * 3 * VDH;    // per-layer qkv_w elems
    const int NQ = VNH * 3 * VDH;                              // 2304
    for (int l = 0; l < VL; ++l) {
        k_ln<<<VM, 256, 0, stream>>>(z, ln1_g + l * VD, ln1_b + l * VD, z1h, 1);
        k_cvt_qkv_t<<<cdiv((long long)VD * NQ, 256), 256, 0, stream>>>(qkv_w + (long long)l * qkvWsz, wh);
        gemm(4, z1h, VD, 0, wh, VD, 0, qkv_b + (long long)l * NQ, nullptr, nullptr, qkvh,
             NQ, 1, 0, 0, VM, NQ, VD, 1.f, 0, 1);
        k_split<<<cdiv((long long)VM * NQ, 256), 256, 0, stream>>>(qkvh, qh, kh, vTh);
        gemm(2, qh, VDH, (long long)VT * VDH, kh, VDH, (long long)VT * VDH,
             nullptr, nullptr, S, nullptr,
             SLD, 1, (long long)SLD * SLD, 0, VT, VT, VDH, 0.125f, 0, (int)BH);
        k_softmax<<<(unsigned)(BH * VT), 256, 0, stream>>>(S, Ph);
        gemm(2, Ph, SLD, (long long)SLD * SLD, vTh, SLD, (long long)VDH * SLD,
             nullptr, nullptr, nullptr, sah,
             VD, VNH, (long long)VT * VD, VDH, VT, VDH, VT, 1.f, 0, (int)BH);
        k_cvt_t<<<cdiv((long long)VD * VD, 256), 256, 0, stream>>>(msa_w + (long long)l * VD * VD, wh, VD, VD);
        gemm(4, sah, VD, 0, wh, VD, 0, msa_b + l * VD, z, z2, nullptr,
             VD, 1, 0, 0, VM, VD, VD, 1.f, 0, 1);
        k_ln<<<VM, 256, 0, stream>>>(z2, ln2_g + l * VD, ln2_b + l * VD, z1h, 1);
        k_cvt_t<<<cdiv((long long)VD * VMLP, 256), 256, 0, stream>>>(w1 + (long long)l * VD * VMLP, wh, VD, VMLP);
        gemm(4, z1h, VD, 0, wh, VD, 0, b1 + (long long)l * VMLP, nullptr, nullptr, h1h,
             VMLP, 1, 0, 0, VM, VMLP, VD, 1.f, 1, 1);
        k_cvt_t<<<cdiv((long long)VMLP * VD, 256), 256, 0, stream>>>(w2 + (long long)l * VMLP * VD, wh, VMLP, VD);
        gemm(4, h1h, VMLP, 0, wh, VMLP, 0, b2 + l * VD, z2, z, nullptr,
             VD, 1, 0, 0, VM, VD, VMLP, 1.f, 1, 1);
    }

    // ---- final LN on cls rows + head ----
    k_ln<<<VB, 256, 0, stream>>>(z, lnf_g, lnf_b, clsh, VT);
    k_cvt_t<<<cdiv((long long)VD * 1000, 256), 256, 0, stream>>>(head_w, wh, VD, 1000);
    gemm(4, clsh, VD, 0, wh, VD, 0, head_b, nullptr, (float*)d_out, nullptr,
         1000, 1, 0, 0, VB, 1000, VD, 1.f, 0, 1);
}
